// CoarseMatching_43877385896316
// MI455X (gfx1250) — compile-verified, matrix operand order checked by reading, not run on hardware
//
#include <hip/hip_runtime.h>
#include <hip/hip_fp16.h>

typedef __attribute__((ext_vector_type(16))) _Float16 v16h;
typedef __attribute__((ext_vector_type(8)))  _Float16 v8h;
typedef __attribute__((ext_vector_type(4)))  _Float16 v4h;
typedef __attribute__((ext_vector_type(8)))  float    v8f;
typedef __attribute__((ext_vector_type(4)))  unsigned int v4u;
typedef __attribute__((ext_vector_type(8)))  int      v8i;
typedef __attribute__((ext_vector_type(4)))  int      v4i;

// Problem constants (match reference setup_inputs)
#define NBATCH 2
#define LDIM   4800
#define SDIM   4800
#define CDIM   256
#define SIM_SCALE  0.0390625f   /* (1/256) / 0.1 */
#define THRV       0.2f

// GEMM blocking: 64x64 output tile per 256-thread block (8 waves),
// each wave computes 16 rows x 32 cols (2 accumulators).
#define BM 64
#define BN 64
// LDS row stride in halves: 256 + 8 pad halves (16B) -> 528B stride,
// bank stride 132 % 64 = 4 -> conflict-free ds_load_b128 row walk.
#define LROW 264

#if __has_builtin(__builtin_amdgcn_tensor_load_to_lds)
#define CM_HAS_TDM 1
#endif

// ---------------------------------------------------------------------------
// One-time f32 -> f16 conversion of the feature tensors into workspace.
// ---------------------------------------------------------------------------
__global__ void cm_cvt_kernel(const float* __restrict__ src,
                              _Float16* __restrict__ dst, int n4) {
    int i = blockIdx.x * blockDim.x + threadIdx.x;
    if (i < n4) {
        float4 x = ((const float4*)src)[i];
        v4h h = { (_Float16)x.x, (_Float16)x.y, (_Float16)x.z, (_Float16)x.w };
        ((v4h*)dst)[i] = h;
    }
}

// ---------------------------------------------------------------------------
// TDM: stage a [nrows x 256] f16 tile (row stride 256) from global into LDS
// with 16B padding every 128 DWORDs (i.e., per 512B row) -> LROW layout.
// D# group0: count=1 | lds_addr | global_addr | type=2
// D# group1: data_size=2B, pad_enable, pad_interval=128DW(code 6),
//            pad_amount=4DW(code 3); tensor_dim0=256, tile_dim0=256,
//            tile_dim1=nrows, tensor_dim0_stride=256.
// ---------------------------------------------------------------------------
#ifdef CM_HAS_TDM
__device__ __forceinline__ void tdm_load_tile(unsigned lds_off,
                                              const _Float16* gptr, int nrows) {
    unsigned long long ga = (unsigned long long)(uintptr_t)gptr;
    v4u g0;
    g0[0] = 1u;                                   // count = 1 (valid user D#)
    g0[1] = lds_off;                              // lds_addr (bytes)
    g0[2] = (unsigned)(ga & 0xffffffffu);         // global_addr[31:0]
    g0[3] = (unsigned)((ga >> 32) & 0x01ffffffu)  // global_addr[56:32]
          | (2u << 30);                           // type = 2 ("image")
    v8i g1;
    g1[0] = (1 << 16)                             // data_size = 2 bytes
          | (1 << 20)                             // pad_enable
          | (6 << 22)                             // pad_interval: 128 DWORDs
          | (3 << 25);                            // pad_amount: 4 DWORDs (16B)
    g1[1] = (CDIM & 0xffff) << 16;                // tensor_dim0 lo16
    g1[2] = ((CDIM >> 16) & 0xffff)               // tensor_dim0 hi16
          | (((NBATCH * LDIM) & 0xffff) << 16);   // tensor_dim1 lo16
    g1[3] = (((NBATCH * LDIM) >> 16) & 0xffff)    // tensor_dim1 hi16
          | ((CDIM & 0xffff) << 16);              // tile_dim0
    g1[4] = nrows & 0xffff;                       // tile_dim1 (tile_dim2 = 0)
    g1[5] = CDIM;                                 // tensor_dim0_stride lo32
    g1[6] = 0;                                    // stride0 hi | stride1 lo
    g1[7] = 0;
    v4i z4 = {0, 0, 0, 0};
#if defined(__clang_major__) && __clang_major__ >= 23
    v8i z8 = {0, 0, 0, 0, 0, 0, 0, 0};
    __builtin_amdgcn_tensor_load_to_lds(g0, g1, z4, z4, z8, 0);
#else
    __builtin_amdgcn_tensor_load_to_lds(g0, g1, z4, z4, 0);
#endif
}
#endif

// Fallback: cooperative copy into the same padded LDS layout.
__device__ __forceinline__ void manual_load_tile(_Float16* tile,
                                                 const _Float16* gptr, int tid) {
    for (int idx = tid; idx < BM * (CDIM / 8); idx += 256) {
        int row = idx >> 5;                 // 32 8-half chunks per row
        int c8  = idx & 31;
        v8h v = *(const v8h*)(gptr + (size_t)row * CDIM + c8 * 8);
        *(v8h*)(tile + row * LROW + c8 * 8) = v;
    }
}

// ---------------------------------------------------------------------------
// Fragment: lane group g (lane>>4): halves 0..7 <- K = k0+8g.. ;
// halves 8..15 <- K = k0+16+8g.. (CDNA5 16-bit A/B layout). Two 16B loads.
// ---------------------------------------------------------------------------
__device__ __forceinline__ v16h load_frag_f16(const _Float16* row, int k0, int g) {
    const _Float16* p = row + k0 + 8 * g;
    v8h lo = *(const v8h*)(p);
    v8h hi = *(const v8h*)(p + 16);
    return __builtin_shufflevector(lo, hi, 0, 1, 2, 3, 4, 5, 6, 7,
                                   8, 9, 10, 11, 12, 13, 14, 15);
}

__global__ void cm_init_kernel(float* p, int n) {
    int i = blockIdx.x * blockDim.x + threadIdx.x;
    if (i < n) p[i] = 0.0f;
}

// ---------------------------------------------------------------------------
// Shared GEMM body: stage 64x256 A/B tiles to LDS (TDM), then 8 K-steps of
// ds_load fragments + 2x v_wmma_f32_16x16x32_f16 per step per wave.
// ---------------------------------------------------------------------------
struct TileAcc { v8f a0, a1; };

__device__ __forceinline__ TileAcc gemm_block(const _Float16* __restrict__ f0h,
                                              const _Float16* __restrict__ f1h,
                                              _Float16* tileA, _Float16* tileB,
                                              int b, int row0blk, int col0blk,
                                              int w, int g, int r) {
    const _Float16* Abase = f0h + ((size_t)b * LDIM + row0blk) * CDIM;
    const _Float16* Bbase = f1h + ((size_t)b * SDIM + col0blk) * CDIM;

#ifdef CM_HAS_TDM
    if (threadIdx.x < 32) {        // wave 0 issues both TDM descriptors
        tdm_load_tile((unsigned)(uintptr_t)(void*)tileA, Abase, BM);
        tdm_load_tile((unsigned)(uintptr_t)(void*)tileB, Bbase, BN);
        __builtin_amdgcn_s_wait_tensorcnt(0);
    }
    __syncthreads();
#else
    manual_load_tile(tileA, Abase, threadIdx.x);
    manual_load_tile(tileB, Bbase, threadIdx.x);
    __syncthreads();
#endif

    const int mr = w >> 1;              // 0..3 : 16-row slice
    const int nc = (w & 1) * 32;        // 0/32 : 32-col slice
    const _Float16* Arow  = tileA + (mr * 16 + r) * LROW;
    const _Float16* Brow0 = tileB + (nc + r) * LROW;
    const _Float16* Brow1 = tileB + (nc + 16 + r) * LROW;

    TileAcc t;
    t.a0 = (v8f){};
    t.a1 = (v8f){};
#pragma unroll
    for (int k0 = 0; k0 < CDIM; k0 += 32) {
        v16h a  = load_frag_f16(Arow, k0, g);
        v16h b0 = load_frag_f16(Brow0, k0, g);
        v16h b1 = load_frag_f16(Brow1, k0, g);
        t.a0 = __builtin_amdgcn_wmma_f32_16x16x32_f16(false, a, false, b0,
                                                      (short)0, t.a0, false, false);
        t.a1 = __builtin_amdgcn_wmma_f32_16x16x32_f16(false, a, false, b1,
                                                      (short)0, t.a1, false, false);
    }
    return t;
}

// ---------------------------------------------------------------------------
// Pass 1: rowsum[l] = sum_s exp(sim), colsum[s] = sum_l exp(sim).
// (No max-subtraction needed: |sim| <~ 5, exp can't overflow; softmax is
//  shift-invariant.)
// ---------------------------------------------------------------------------
__global__ void cm_sim_sums_kernel(const _Float16* __restrict__ f0h,
                                   const _Float16* __restrict__ f1h,
                                   float* __restrict__ rowsum,
                                   float* __restrict__ colsum) {
    __shared__ _Float16 tileA[BM * LROW];
    __shared__ _Float16 tileB[BN * LROW];

    const int b    = blockIdx.z;
    const int lane = threadIdx.x & 31;
    const int w    = threadIdx.x >> 5;
    const int g    = lane >> 4;
    const int r    = lane & 15;
    const int row0blk = blockIdx.y * BM;
    const int col0blk = blockIdx.x * BN;

    TileAcc t = gemm_block(f0h, f1h, tileA, tileB, b, row0blk, col0blk, w, g, r);

    const int row0 = row0blk + (w >> 1) * 16;     // wave's 16-row slice
    const int c0   = col0blk + (w & 1) * 32;      // wave's first 16-col tile

    float ev0[8], ev1[8];
    float ca0 = 0.0f, ca1 = 0.0f;
#pragma unroll
    for (int v = 0; v < 8; ++v) {
        float e0 = __expf(t.a0[v] * SIM_SCALE);
        float e1 = __expf(t.a1[v] * SIM_SCALE);
        ev0[v] = e0; ev1[v] = e1;
        ca0 += e0;  ca1 += e1;
    }
    ca0 += __shfl_xor(ca0, 16, 32);
    ca1 += __shfl_xor(ca1, 16, 32);
    if (g == 0) {
        atomicAdd(&colsum[(size_t)b * SDIM + c0 + r],      ca0);
        atomicAdd(&colsum[(size_t)b * SDIM + c0 + 16 + r], ca1);
    }
#pragma unroll
    for (int v = 0; v < 8; ++v) {
        float rs = ev0[v] + ev1[v];
#pragma unroll
        for (int d = 1; d < 16; d <<= 1) rs += __shfl_xor(rs, d, 32);
        if (r == 0)
            atomicAdd(&rowsum[(size_t)b * LDIM + row0 + v + 8 * g], rs);
    }
}

// ---------------------------------------------------------------------------
// Pass 2: conf = exp(sim)^2 / (rowsum*colsum); track row/col conf maxima
// (int-punned atomicMax; conf >= 0).
// ---------------------------------------------------------------------------
__global__ void cm_conf_kernel(const _Float16* __restrict__ f0h,
                               const _Float16* __restrict__ f1h,
                               const float* __restrict__ rowsum,
                               const float* __restrict__ colsum,
                               float* __restrict__ conf,
                               int* __restrict__ rcmax,
                               int* __restrict__ ccmax) {
    __shared__ _Float16 tileA[BM * LROW];
    __shared__ _Float16 tileB[BN * LROW];

    const int b    = blockIdx.z;
    const int lane = threadIdx.x & 31;
    const int w    = threadIdx.x >> 5;
    const int g    = lane >> 4;
    const int r    = lane & 15;
    const int row0blk = blockIdx.y * BM;
    const int col0blk = blockIdx.x * BN;

    TileAcc t = gemm_block(f0h, f1h, tileA, tileB, b, row0blk, col0blk, w, g, r);

    const int row0 = row0blk + (w >> 1) * 16;
    const int c0   = col0blk + (w & 1) * 32;

    const float inv_cs0 = 1.0f / colsum[(size_t)b * SDIM + c0 + r];
    const float inv_cs1 = 1.0f / colsum[(size_t)b * SDIM + c0 + 16 + r];

    float cmax0 = 0.0f, cmax1 = 0.0f;
#pragma unroll
    for (int v = 0; v < 8; ++v) {
        const int m = row0 + v + 8 * g;
        const float inv_rs = 1.0f / rowsum[(size_t)b * LDIM + m];
        float e0 = __expf(t.a0[v] * SIM_SCALE);
        float e1 = __expf(t.a1[v] * SIM_SCALE);
        float c0v = (e0 * e0) * inv_cs0 * inv_rs;
        float c1v = (e1 * e1) * inv_cs1 * inv_rs;
        conf[((size_t)b * LDIM + m) * SDIM + c0 + r]      = c0v;
        conf[((size_t)b * LDIM + m) * SDIM + c0 + 16 + r] = c1v;
        cmax0 = fmaxf(cmax0, c0v);
        cmax1 = fmaxf(cmax1, c1v);

        float rm = fmaxf(c0v, c1v);
#pragma unroll
        for (int d = 1; d < 16; d <<= 1) rm = fmaxf(rm, __shfl_xor(rm, d, 32));
        if (r == 0)
            atomicMax(&rcmax[(size_t)b * LDIM + m], __float_as_int(rm));
    }
    cmax0 = fmaxf(cmax0, __shfl_xor(cmax0, 16, 32));
    cmax1 = fmaxf(cmax1, __shfl_xor(cmax1, 16, 32));
    if (g == 0) {
        atomicMax(&ccmax[(size_t)b * SDIM + c0 + r],      __float_as_int(cmax0));
        atomicMax(&ccmax[(size_t)b * SDIM + c0 + 16 + r], __float_as_int(cmax1));
    }
}

// ---------------------------------------------------------------------------
// Pass 3: match extraction. One block per (b, l) row.
// mask = conf>thr & border_valid & conf==rowmax & conf==colmax
// j_ids = first True column (0 if none); match_mask = any; mconf = conf or 0.
// ---------------------------------------------------------------------------
__global__ void cm_match_kernel(const float* __restrict__ conf,
                                const float* __restrict__ rcmaxf,
                                const float* __restrict__ ccmaxf,
                                const int* __restrict__ w0p,
                                const int* __restrict__ w1p,
                                float* __restrict__ out_mm,
                                float* __restrict__ out_jid,
                                float* __restrict__ out_mc) {
    const int l = blockIdx.x;
    const int b = blockIdx.y;
    const int w0 = *w0p;
    const int w1 = *w1p;

    const float rm = rcmaxf[(size_t)b * LDIM + l];
    const bool  v0 = ((l / w0) >= 2) && ((l % w0) >= 2);

    int best = 0x7fffffff;
    if (v0 && rm > THRV) {
        const float* crow = conf + ((size_t)b * LDIM + l) * SDIM;
        for (int s = threadIdx.x; s < SDIM; s += blockDim.x) {
            float c = crow[s];
            if (c > THRV && c == rm && c == ccmaxf[(size_t)b * SDIM + s]) {
                if (((s / w1) >= 2) && ((s % w1) >= 2) && s < best) best = s;
            }
        }
    }

    __shared__ int red[256];
    red[threadIdx.x] = best;
    __syncthreads();
    for (int off = 128; off > 0; off >>= 1) {
        if (threadIdx.x < off)
            red[threadIdx.x] = min(red[threadIdx.x], red[threadIdx.x + off]);
        __syncthreads();
    }

    if (threadIdx.x == 0) {
        int  bi    = red[0];
        bool found = (bi != 0x7fffffff);
        int  j     = found ? bi : 0;
        size_t o   = (size_t)b * LDIM + l;
        out_mm[o]  = found ? 1.0f : 0.0f;
        out_jid[o] = (float)j;
        out_mc[o]  = found ? conf[o * SDIM + j] : 0.0f;
    }
}

extern "C" void kernel_launch(void* const* d_in, const int* in_sizes, int n_in,
                              void* d_out, int out_size, void* d_ws, size_t ws_size,
                              hipStream_t stream) {
    (void)in_sizes; (void)n_in; (void)out_size; (void)ws_size;

    const float* f0  = (const float*)d_in[0];   // [N, L, C] f32
    const float* f1  = (const float*)d_in[1];   // [N, S, C] f32
    const int*   w0p = (const int*)d_in[3];     // w0c
    const int*   w1p = (const int*)d_in[5];     // w1c

    // Workspace layout:
    //   floats: rowsum[N*L] | colsum[N*S] | rcmax[N*L] | ccmax[N*S]   (153.6 KB)
    //   then:   f0h[N*L*C] f16 | f1h[N*S*C] f16                      (9.8 MB)
    float* ws     = (float*)d_ws;
    float* rowsum = ws;
    float* colsum = rowsum + (size_t)NBATCH * LDIM;
    float* rcmax  = colsum + (size_t)NBATCH * SDIM;
    float* ccmax  = rcmax  + (size_t)NBATCH * LDIM;
    const int ws_elems = 2 * NBATCH * (LDIM + SDIM);

    _Float16* f0h = (_Float16*)(ccmax + (size_t)NBATCH * SDIM);
    _Float16* f1h = f0h + (size_t)NBATCH * LDIM * CDIM;

    // Output layout (floats): conf[N*L*S] | match_mask[N*L] | j_ids[N*L] | mconf[N*L]
    float* conf = (float*)d_out;
    float* mm   = conf + (size_t)NBATCH * LDIM * SDIM;
    float* jid  = mm   + (size_t)NBATCH * LDIM;
    float* mc   = jid  + (size_t)NBATCH * LDIM;

    cm_init_kernel<<<(ws_elems + 255) / 256, 256, 0, stream>>>(ws, ws_elems);

    const int n4 = NBATCH * LDIM * CDIM / 4;    // 614400 float4 groups
    cm_cvt_kernel<<<(n4 + 255) / 256, 256, 0, stream>>>(f0, f0h, n4);
    cm_cvt_kernel<<<(n4 + 255) / 256, 256, 0, stream>>>(f1, f1h, n4);

    dim3 gemm_grid(SDIM / BN, LDIM / BM, NBATCH);   // 75 x 75 x 2
    cm_sim_sums_kernel<<<gemm_grid, 256, 0, stream>>>(f0h, f1h, rowsum, colsum);
    cm_conf_kernel<<<gemm_grid, 256, 0, stream>>>(f0h, f1h, rowsum, colsum, conf,
                                                  (int*)rcmax, (int*)ccmax);

    dim3 match_grid(LDIM, NBATCH);
    cm_match_kernel<<<match_grid, 256, 0, stream>>>(conf, rcmax, ccmax,
                                                    w0p, w1p, mm, jid, mc);
}
